// GGNN_27084063769018
// MI455X (gfx1250) — matrix operand order repaired and checked by reading, not compiled, not run on hardware
//
#include <hip/hip_runtime.h>

// ---------------------------------------------------------------------------
// Types for CDNA5 WMMA (wave32): v_wmma_f32_16x16x32_bf16
// ---------------------------------------------------------------------------
typedef __attribute__((ext_vector_type(16))) __bf16         bf16x16;
typedef __attribute__((ext_vector_type(8)))  float          v8f;
typedef __attribute__((ext_vector_type(8)))  unsigned short ushort8;

union FragU {
    bf16x16 v;
    ushort8 h[2];
};

#define D_DIM 128

// round-to-nearest-even f32 -> bf16 bits
__device__ __forceinline__ unsigned short f2bf(float x) {
    unsigned int u = __float_as_uint(x);
    unsigned int r = u + 0x7FFFu + ((u >> 16) & 1u);
    return (unsigned short)(r >> 16);
}
__device__ __forceinline__ float bf2f(unsigned short b) {
    return __uint_as_float(((unsigned int)b) << 16);
}

// ---------------------------------------------------------------------------
// gfx1250 async global->LDS staging (ASYNCcnt path), guarded with fallback
// ---------------------------------------------------------------------------
#if defined(__has_builtin)
#if __has_builtin(__builtin_amdgcn_global_load_async_to_lds_b128)
#define HAVE_ASYNC_LDS 1
#endif
#endif

// builtin operand types: <4 x i32>* in AS(1) (src) and AS(3) (dst)
typedef int v4i_g __attribute__((vector_size(16)));
typedef __attribute__((address_space(1))) v4i_g glob_v4i;
typedef __attribute__((address_space(3))) v4i_g lds_v4i;

__device__ __forceinline__ void wait_async0() {
#if defined(__has_builtin) && __has_builtin(__builtin_amdgcn_s_wait_asynccnt)
    __builtin_amdgcn_s_wait_asynccnt(0);
#else
    asm volatile("s_wait_asynccnt 0" ::: "memory");
#endif
}

// copy 16 bytes (8 bf16) global -> LDS, per lane
__device__ __forceinline__ void copy16_g2lds(const unsigned short* g, unsigned short* l) {
#ifdef HAVE_ASYNC_LDS
    __builtin_amdgcn_global_load_async_to_lds_b128(
        (glob_v4i*)(unsigned long long)(__SIZE_TYPE__)g,
        (lds_v4i*)(unsigned int)(__SIZE_TYPE__)l,
        0, 0);
#else
    *(ushort8*)l = *(const ushort8*)g;
#endif
}

// ---------------------------------------------------------------------------
// WMMA fragment loaders
// ---------------------------------------------------------------------------
// A-matrix fragment (16x32 bf16, MxK) from LDS tile sh[16][128] (bf16 bits).
__device__ __forceinline__ bf16x16 load_a_frag(const unsigned short* sh, int lane, int kk) {
    const int row  = lane & 15;
    const int off0 = (lane < 16) ? 0 : 8;
    FragU f;
    f.h[0] = *(const ushort8*)(sh + row * D_DIM + kk + off0);
    f.h[1] = *(const ushort8*)(sh + row * D_DIM + kk + 16 + off0);
    return f.v;
}

// B-matrix fragment (32x16 bf16, KxN). B stored [c][k]; two 8-bf16 chunks/lane (k = 8h.., 16+8h..).
__device__ __forceinline__ bf16x16 load_b_frag(const unsigned short* Bp, int lane, int kk, int c0) {
    const int c    = c0 + (lane & 15);
    const int koff = kk + ((lane < 16) ? 0 : 8);
    FragU f;
    f.h[0] = *(const ushort8*)(Bp + (size_t)c * D_DIM + koff);
    f.h[1] = *(const ushort8*)(Bp + (size_t)c * D_DIM + koff + 16);
    return f.v;
}

// ---------------------------------------------------------------------------
// Prep kernels (once per launch)
// ---------------------------------------------------------------------------
__global__ void k_f32_to_bf16(const float* __restrict__ src, unsigned short* __restrict__ dst, int n) {
    int i = blockIdx.x * blockDim.x + threadIdx.x;
    if (i < n) dst[i] = f2bf(src[i]);
}

// weight[l][k][c] (m = h @ W) -> WT[l][c][k]  (K-contiguous for WMMA B operand)
__global__ void k_transpose_bf16(const float* __restrict__ W, unsigned short* __restrict__ WT, int total) {
    int i = blockIdx.x * blockDim.x + threadIdx.x;
    if (i >= total) return;
    int l = i >> 14;
    int rem = i & 16383;
    int c = rem >> 7, k = rem & 127;
    WT[i] = f2bf(W[l * 16384 + k * D_DIM + c]);
}

__global__ void k_zero_f32(float* __restrict__ p, int n4) {
    int i = blockIdx.x * blockDim.x + threadIdx.x;
    if (i < n4) ((float4*)p)[i] = make_float4(0.f, 0.f, 0.f, 0.f);
}

// ---------------------------------------------------------------------------
// Projection GEMM: h = x @ lin_w^T + b.  A f32 (converted on stage),
// writes h in f32 AND bf16 (persistent bf16 copy for later layers).
// ---------------------------------------------------------------------------
__global__ void k_gemm_proj(const float* __restrict__ A, const unsigned short* __restrict__ B,
                            const float* __restrict__ bias, float* __restrict__ C,
                            unsigned short* __restrict__ Cbf) {
    __shared__ unsigned short a_sh[16 * D_DIM];
    const int m0  = blockIdx.x * 16;
    const int tid = threadIdx.x;

    for (int i = tid; i < 16 * D_DIM; i += 256)
        a_sh[i] = f2bf(A[(size_t)(m0 + (i >> 7)) * D_DIM + (i & 127)]);
    __syncthreads();

    const int wave = tid >> 5, lane = tid & 31;
    const int c0 = wave * 16;

    v8f acc = {};
    #pragma unroll
    for (int kk = 0; kk < D_DIM; kk += 32) {
        bf16x16 a = load_a_frag(a_sh, lane, kk);
        bf16x16 b = load_b_frag(B, lane, kk, c0);
        acc = __builtin_amdgcn_wmma_f32_16x16x32_bf16(false, a, false, b, (short)0, acc, false, false);
    }

    const int mbase = (lane >> 4) * 8;
    const int N     = lane & 15;
    const float bv  = bias[c0 + N];
    #pragma unroll
    for (int r = 0; r < 8; ++r) {
        const float v = acc[r] + bv;
        const size_t idx = (size_t)(m0 + mbase + r) * D_DIM + c0 + N;
        C[idx]   = v;
        Cbf[idx] = f2bf(v);
    }
}

// ---------------------------------------------------------------------------
// Message GEMM: m_bf16 = h_bf16 @ weight[l].  A already bf16 -> async-staged
// straight into LDS (no VGPR round trip).  Output bf16 (halves scatter traffic).
// ---------------------------------------------------------------------------
__global__ void k_gemm_msg(const unsigned short* __restrict__ Abf,
                           const unsigned short* __restrict__ B,
                           unsigned short* __restrict__ Cbf) {
    __shared__ unsigned short a_sh[16 * D_DIM];
    const int m0  = blockIdx.x * 16;
    const int tid = threadIdx.x;

    {   // 16 rows x 16 chunks of 8 bf16 (16B) = 4KB tile; one b128 per thread
        const int r = tid >> 4, ch = (tid & 15) * 8;
        copy16_g2lds(Abf + (size_t)(m0 + r) * D_DIM + ch, a_sh + r * D_DIM + ch);
#ifdef HAVE_ASYNC_LDS
        wait_async0();
#endif
    }
    __syncthreads();

    const int wave = tid >> 5, lane = tid & 31;
    const int c0 = wave * 16;

    v8f acc = {};
    #pragma unroll
    for (int kk = 0; kk < D_DIM; kk += 32) {
        bf16x16 a = load_a_frag(a_sh, lane, kk);
        bf16x16 b = load_b_frag(B, lane, kk, c0);
        acc = __builtin_amdgcn_wmma_f32_16x16x32_bf16(false, a, false, b, (short)0, acc, false, false);
    }

    const int mbase = (lane >> 4) * 8;
    const int N     = lane & 15;
    #pragma unroll
    for (int r = 0; r < 8; ++r)
        Cbf[(size_t)(m0 + mbase + r) * D_DIM + c0 + N] = f2bf(acc[r]);
}

// ---------------------------------------------------------------------------
// Scatter-add (segment_sum): 16 lanes/edge, 8 bf16 channels each (one 16B
// gather per lane), f32 accumulation with hardware atomics.
// ---------------------------------------------------------------------------
__global__ void k_scatter_add(const unsigned short* __restrict__ m, const long long* __restrict__ ei,
                              float* __restrict__ agg, int E) {
    int gid = blockIdx.x * blockDim.x + threadIdx.x;
    int e = gid >> 4;
    if (e >= E) return;
    int part = (gid & 15) * 8;
    long long s = ei[e];
    long long d = ei[(size_t)E + e];
    ushort8 v = *(const ushort8*)(m + (size_t)s * D_DIM + part);
    float* ap = agg + (size_t)d * D_DIM + part;
    #pragma unroll
    for (int j = 0; j < 8; ++j)
        __hip_atomic_fetch_add(ap + j, bf2f(v[j]), __ATOMIC_RELAXED, __HIP_MEMORY_SCOPE_AGENT);
}

// ---------------------------------------------------------------------------
// Fused GRU: gi = agg@w_ih^T, gh = h@w_hh^T (WMMA), gates, h updated in place
// (f32 in d_out + bf16 shadow).  gi/gh live only in LDS.
// ---------------------------------------------------------------------------
__global__ void k_gru_fused(const float* __restrict__ agg, float* __restrict__ h,
                            unsigned short* __restrict__ hbf,
                            const unsigned short* __restrict__ wih,
                            const unsigned short* __restrict__ whh,
                            const float* __restrict__ bih, const float* __restrict__ bhh) {
    __shared__ unsigned short agg_sh[16 * D_DIM];   // 4 KB
    __shared__ unsigned short h_sh[16 * D_DIM];     // 4 KB
    __shared__ float gi_sh[16 * 384];               // 24 KB
    __shared__ float gh_sh[16 * 384];               // 24 KB

    const int m0  = blockIdx.x * 16;
    const int tid = threadIdx.x;

    {   // async-stage h tile (already bf16) while converting agg f32->bf16
        const int r = tid >> 4, ch = (tid & 15) * 8;
        copy16_g2lds(hbf + (size_t)(m0 + r) * D_DIM + ch, h_sh + r * D_DIM + ch);
    }
    for (int i = tid; i < 16 * D_DIM; i += 256)
        agg_sh[i] = f2bf(agg[(size_t)(m0 + (i >> 7)) * D_DIM + (i & 127)]);
#ifdef HAVE_ASYNC_LDS
    wait_async0();
#endif
    __syncthreads();

    const int wave = tid >> 5, lane = tid & 31;
    #pragma unroll
    for (int t = 0; t < 6; ++t) {
        const int tt = wave + 8 * t;                         // 0..47
        const bool is_gi = (tt < 24);
        const unsigned short* Bp  = is_gi ? wih : whh;
        const unsigned short* Ash = is_gi ? agg_sh : h_sh;
        float* Osh = is_gi ? gi_sh : gh_sh;
        const int cb = is_gi ? tt : tt - 24;
        const int c0 = cb * 16;

        v8f acc = {};
        #pragma unroll
        for (int kk = 0; kk < D_DIM; kk += 32) {
            bf16x16 a = load_a_frag(Ash, lane, kk);
            bf16x16 b = load_b_frag(Bp, lane, kk, c0);
            acc = __builtin_amdgcn_wmma_f32_16x16x32_bf16(false, a, false, b, (short)0, acc, false, false);
        }
        const int mbase = (lane >> 4) * 8;
        const int N     = lane & 15;
        #pragma unroll
        for (int r = 0; r < 8; ++r)
            Osh[(mbase + r) * 384 + c0 + N] = acc[r];
    }
    __syncthreads();

    for (int i = tid; i < 16 * D_DIM; i += 256) {
        const int r = i >> 7, c = i & 127;
        const float gir = gi_sh[r * 384 + c]       + bih[c];
        const float giz = gi_sh[r * 384 + 128 + c] + bih[128 + c];
        const float gin = gi_sh[r * 384 + 256 + c] + bih[256 + c];
        const float ghr = gh_sh[r * 384 + c]       + bhh[c];
        const float ghz = gh_sh[r * 384 + 128 + c] + bhh[128 + c];
        const float ghn = gh_sh[r * 384 + 256 + c] + bhh[256 + c];
        const float rg = 1.f / (1.f + __expf(-(gir + ghr)));
        const float zg = 1.f / (1.f + __expf(-(giz + ghz)));
        const float ng = tanhf(gin + rg * ghn);
        const size_t idx = (size_t)(m0 + r) * D_DIM + c;
        const float hold = h[idx];
        const float hnew = (1.f - zg) * ng + zg * hold;
        h[idx]   = hnew;
        hbf[idx] = f2bf(hnew);
    }
}

// ---------------------------------------------------------------------------
// Host-side orchestration
// ---------------------------------------------------------------------------
extern "C" void kernel_launch(void* const* d_in, const int* in_sizes, int n_in,
                              void* d_out, int out_size, void* d_ws, size_t ws_size,
                              hipStream_t stream) {
    const float* x     = (const float*)d_in[0];   // [N,128]
    const float* lin_w = (const float*)d_in[1];   // [128,128]
    const float* lin_b = (const float*)d_in[2];   // [128]
    const float* W     = (const float*)d_in[3];   // [L,128,128]
    const float* w_ih  = (const float*)d_in[4];   // [384,128]
    const float* w_hh  = (const float*)d_in[5];   // [384,128]
    const float* b_ih  = (const float*)d_in[6];   // [384]
    const float* b_hh  = (const float*)d_in[7];   // [384]
    const long long* edge_index = (const long long*)d_in[8];   // [2,E] int64

    const int N = in_sizes[0] / D_DIM;
    const int L = in_sizes[3] / (D_DIM * D_DIM);
    const int E = in_sizes[8] / 2;

    float* h = (float*)d_out;                     // f32 h lives in d_out throughout

    // workspace layout
    char* ws = (char*)d_ws;
    size_t off = 0;
    float* agg           = (float*)(ws + off);          off += (size_t)N * D_DIM * sizeof(float);
    unsigned short* h_bf = (unsigned short*)(ws + off); off += (size_t)N * D_DIM * 2;
    unsigned short* m_bf = (unsigned short*)(ws + off); off += (size_t)N * D_DIM * 2;
    unsigned short* linw_bf = (unsigned short*)(ws + off); off += (size_t)D_DIM * D_DIM * 2;
    unsigned short* wih_bf  = (unsigned short*)(ws + off); off += (size_t)3 * D_DIM * D_DIM * 2;
    unsigned short* whh_bf  = (unsigned short*)(ws + off); off += (size_t)3 * D_DIM * D_DIM * 2;
    unsigned short* wT_bf   = (unsigned short*)(ws + off); off += (size_t)L * D_DIM * D_DIM * 2;
    (void)ws_size; (void)n_in; (void)out_size;

    // --- weight prep (bf16, WMMA-B layout) ---
    {
        int n = D_DIM * D_DIM;
        k_f32_to_bf16<<<(n + 255) / 256, 256, 0, stream>>>(lin_w, linw_bf, n);
        n = 3 * D_DIM * D_DIM;
        k_f32_to_bf16<<<(n + 255) / 256, 256, 0, stream>>>(w_ih, wih_bf, n);
        k_f32_to_bf16<<<(n + 255) / 256, 256, 0, stream>>>(w_hh, whh_bf, n);
        int t = L * D_DIM * D_DIM;
        k_transpose_bf16<<<(t + 255) / 256, 256, 0, stream>>>(W, wT_bf, t);
    }

    const int row_blocks = N / 16;                // N = 100000 = 16 * 6250

    // --- input projection: h = x @ lin_w^T + lin_b  (f32 + bf16 shadow) ---
    k_gemm_proj<<<row_blocks, 256, 0, stream>>>(x, linw_bf, lin_b, h, h_bf);

    // --- L gated message-passing steps ---
    const int zero_blocks = ((N * D_DIM) / 4 + 255) / 256;
    const int scat_blocks = (E * 16 + 255) / 256;
    for (int l = 0; l < L; ++l) {
        // m = h @ weight[l]  (bf16 in, bf16 out)
        k_gemm_msg<<<row_blocks, 256, 0, stream>>>(h_bf, wT_bf + (size_t)l * D_DIM * D_DIM, m_bf);
        // agg = segment_sum(m[src], dst)  (f32 accumulation)
        k_zero_f32<<<zero_blocks, 256, 0, stream>>>(agg, (N * D_DIM) / 4);
        k_scatter_add<<<scat_blocks, 256, 0, stream>>>(m_bf, edge_index, agg, E);
        // h = GRU(agg, h)  (in place, refreshes bf16 shadow)
        k_gru_fused<<<row_blocks, 256, 0, stream>>>(agg, h, h_bf, wih_bf, whh_bf, b_ih, b_hh);
    }
}
